// EncoderLayer_72971494359423
// MI455X (gfx1250) — matrix operand order repaired and checked by reading, not compiled, not run on hardware
//
#include <hip/hip_runtime.h>
#include <hip/hip_bf16.h>

typedef __bf16 bf16;
typedef __attribute__((ext_vector_type(16))) __bf16 bf16x16;
typedef __attribute__((ext_vector_type(8)))  __bf16 bf16x8;
typedef __attribute__((ext_vector_type(8)))  float  floatx8;

#define B_ 4
#define N_ 2048
#define C_ 512
#define H_ 8
#define HD_ 64
#define HIDE_ 2048

__device__ inline floatx8 wmma_bf16(bf16x16 a, bf16x16 b, floatx8 c) {
    return __builtin_amdgcn_wmma_f32_16x16x32_bf16(
        /*neg_a=*/false, a, /*neg_b=*/false, b,
        /*c_mod=*/(short)0, c, /*reuse_a=*/false, /*reuse_b=*/false);
}

// A fragment (16x32, MxK) from row-major tile, ld in elements.
__device__ inline bf16x16 load_a_frag(const bf16* p, int ld, int c0, int lane) {
    int m = lane & 15, half = lane >> 4;
    const bf16* base = p + (size_t)m * ld + c0 + half * 8;
    union { bf16x16 v; bf16x8 h[2]; } u;
    u.h[0] = *(const bf16x8*)(base);
    u.h[1] = *(const bf16x8*)(base + 16);
    return u.v;
}

// B fragment (32x16, KxN): B[c][n] = T[n0+n][c] for row-major tile T.
__device__ inline bf16x16 load_b_frag(const bf16* p, int ld, int n0, int c0, int lane) {
    int n = lane & 15, half = lane >> 4;
    const bf16* base = p + (size_t)(n0 + n) * ld + c0 + half * 8;
    union { bf16x16 v; bf16x8 h[2]; } u;
    u.h[0] = *(const bf16x8*)(base);
    u.h[1] = *(const bf16x8*)(base + 16);
    return u.v;
}

// Async 16B global -> LDS copy (per-lane addresses), tracked with ASYNCcnt.
// GVS addressing: SGPR-pair base + signed 32-bit per-lane byte offset.
__device__ inline void async_b128(unsigned lds_off, unsigned goff, const void* base) {
    asm volatile("global_load_async_to_lds_b128 %0, %1, %2"
                 :: "v"(lds_off), "v"(goff), "s"(base) : "memory");
}

// ---------------- fp32 -> bf16 convert (optionally scaled) ----------------
__global__ __launch_bounds__(256) void cvt_kernel(const float* __restrict__ in,
                                                  bf16* __restrict__ out,
                                                  float scale, int n4) {
    int i = blockIdx.x * 256 + threadIdx.x;
    if (i < n4) {
        float4 f = ((const float4*)in)[i];
        union { bf16 b[4]; uint2 u; } r;
        r.b[0] = (bf16)(f.x * scale);
        r.b[1] = (bf16)(f.y * scale);
        r.b[2] = (bf16)(f.z * scale);
        r.b[3] = (bf16)(f.w * scale);
        *(uint2*)(out + 4 * (size_t)i) = r.u;
    }
}

// ---------------- flash attention ----------------
// grid: (N/64, H, B), block: 128 threads (4 waves); wave w owns q-rows [qb0+16w, +16)
__global__ __launch_bounds__(128) void attn_kernel(const bf16* __restrict__ qb,
                                                   const bf16* __restrict__ kb,
                                                   const bf16* __restrict__ vb,
                                                   float* __restrict__ attn) {
    int qb0 = blockIdx.x * 64;
    int h = blockIdx.y, b = blockIdx.z;
    int wave = threadIdx.x >> 5, lane = threadIdx.x & 31;

    __shared__ alignas(16) bf16 kt[64 * 64];        // K tile, row-major [krow][hd]
    __shared__ alignas(16) bf16 vt[64 * 64];        // V tile transposed [hd][krow]
    __shared__ alignas(16) bf16 pt[4][16 * 64];     // per-wave P tile [16][64]
    unsigned kt_lds = (unsigned)(uintptr_t)&kt[0];

    int qrow = qb0 + wave * 16;
    bf16x16 qa0, qa1;
    {
        const bf16* qptr = qb + ((size_t)b * N_ + qrow) * C_ + h * HD_;
        qa0 = load_a_frag(qptr, C_, 0, lane);
        qa1 = load_a_frag(qptr, C_, 32, lane);
    }

    floatx8 o[4] = {{}, {}, {}, {}};
    float rm[8], rl[8];
#pragma unroll
    for (int r = 0; r < 8; ++r) { rm[r] = -1e30f; rl[r] = 0.f; }

    for (int kb0 = 0; kb0 < N_; kb0 += 64) {
        __syncthreads();   // all waves done reading previous tiles
        const bf16* ksrc = kb + ((size_t)b * N_ + kb0) * C_ + h * HD_;
#pragma unroll
        for (int j = 0; j < 4; ++j) {       // K tile: 512 16B vectors / 128 threads
            int i = threadIdx.x + j * 128;
            int row = i >> 3, c8 = (i & 7) * 8;
            async_b128(kt_lds + (unsigned)(row * 64 + c8) * 2,
                       (unsigned)((row * C_ + c8) * 2), ksrc);
        }
        const bf16* vsrc = vb + ((size_t)b * N_ + kb0) * C_ + h * HD_;
#pragma unroll
        for (int j = 0; j < 4; ++j) {       // V tile transposed, 8 bf16 per step
            int i = threadIdx.x + j * 128;
            int row = i >> 3, c8 = (i & 7) * 8;
            union { uint4 u; bf16 e[8]; } d;
            d.u = *(const uint4*)&vsrc[(size_t)row * C_ + c8];
#pragma unroll
            for (int x = 0; x < 8; ++x) vt[(c8 + x) * 64 + row] = d.e[x];
        }
        if (kb0 + 64 < N_)                   // warm next V-block (global_prefetch_b8)
            __builtin_prefetch(vsrc + (size_t)64 * C_ + threadIdx.x * 16, 0, 0);
        asm volatile("s_wait_asynccnt 0x0" ::: "memory");
        __syncthreads();

        // S = Q * K^T, 4 16x16 tiles, contraction over hd=64
        floatx8 s[4];
#pragma unroll
        for (int t = 0; t < 4; ++t) {
            floatx8 acc = {};
            acc = wmma_bf16(qa0, load_b_frag(kt, 64, t * 16, 0, lane), acc);
            acc = wmma_bf16(qa1, load_b_frag(kt, 64, t * 16, 32, lane), acc);
            s[t] = acc;
        }

        float nm[8];
#pragma unroll
        for (int r = 0; r < 8; ++r) {
            float v = fmaxf(fmaxf(s[0][r], s[1][r]), fmaxf(s[2][r], s[3][r]));
            v = fmaxf(v, __shfl_xor(v, 1, 32));
            v = fmaxf(v, __shfl_xor(v, 2, 32));
            v = fmaxf(v, __shfl_xor(v, 4, 32));
            v = fmaxf(v, __shfl_xor(v, 8, 32));
            nm[r] = fmaxf(rm[r], v);
        }

        bf16* myp = pt[wave];
        float rs[8];
#pragma unroll
        for (int r = 0; r < 8; ++r) rs[r] = 0.f;
#pragma unroll
        for (int t = 0; t < 4; ++t) {
#pragma unroll
            for (int r = 0; r < 8; ++r) {
                float p = __expf(s[t][r] - nm[r]);
                rs[r] += p;
                int m = r + (lane >> 4) * 8;
                myp[m * 64 + t * 16 + (lane & 15)] = (bf16)p;
            }
        }
#pragma unroll
        for (int r = 0; r < 8; ++r) {
            float v = rs[r];
            v += __shfl_xor(v, 1, 32);
            v += __shfl_xor(v, 2, 32);
            v += __shfl_xor(v, 4, 32);
            v += __shfl_xor(v, 8, 32);
            float sc = __expf(rm[r] - nm[r]);
            rl[r] = rl[r] * sc + v;
            rm[r] = nm[r];
#pragma unroll
            for (int t = 0; t < 4; ++t) o[t][r] *= sc;
        }

        asm volatile("s_wait_dscnt 0x0" ::: "memory");  // P stores visible to our loads

        bf16x16 pa0 = load_a_frag(myp, 64, 0, lane);
        bf16x16 pa1 = load_a_frag(myp, 64, 32, lane);
#pragma unroll
        for (int t = 0; t < 4; ++t) {
            o[t] = wmma_bf16(pa0, load_b_frag(vt, 64, t * 16, 0, lane), o[t]);
            o[t] = wmma_bf16(pa1, load_b_frag(vt, 64, t * 16, 32, lane), o[t]);
        }
    }

    float* outbase = attn + ((size_t)b * N_ + qrow) * C_ + h * HD_;
#pragma unroll
    for (int t = 0; t < 4; ++t) {
#pragma unroll
        for (int r = 0; r < 8; ++r) {
            int m = r + (lane >> 4) * 8;
            int n = t * 16 + (lane & 15);
            outbase[(size_t)m * C_ + n] = o[t][r] / rl[r];
        }
    }
}

// ---------------- layernorm (optional residual add; fp32 and/or bf16 outputs) --------
__global__ __launch_bounds__(256) void ln_kernel(const float* __restrict__ x,
                                                 const float* __restrict__ addend,
                                                 const float* __restrict__ w,
                                                 const float* __restrict__ bias,
                                                 float* __restrict__ outf,
                                                 bf16* __restrict__ outb) {
    int row = blockIdx.x;
    const float* xr = x + (size_t)row * C_;
    int t = threadIdx.x;
    float v[2], s = 0.f, s2 = 0.f;
#pragma unroll
    for (int i = 0; i < 2; ++i) {
        int c = t + i * 256;
        float val = xr[c];
        if (addend) val += addend[(size_t)row * C_ + c];
        v[i] = val;
        s += val;
        s2 += val * val;
    }
    __shared__ float red[8], red2[8];
    for (int off = 16; off; off >>= 1) {
        s  += __shfl_xor(s, off, 32);
        s2 += __shfl_xor(s2, off, 32);
    }
    int wave = t >> 5, lane = t & 31;
    if (lane == 0) { red[wave] = s; red2[wave] = s2; }
    __syncthreads();
    if (wave == 0) {
        s  = (lane < 8) ? red[lane]  : 0.f;
        s2 = (lane < 8) ? red2[lane] : 0.f;
        for (int off = 4; off; off >>= 1) {
            s  += __shfl_xor(s, off, 32);
            s2 += __shfl_xor(s2, off, 32);
        }
        if (lane == 0) { red[0] = s; red2[0] = s2; }
    }
    __syncthreads();
    float mean = red[0] * (1.f / C_);
    float var  = red2[0] * (1.f / C_) - mean * mean;
    float inv  = rsqrtf(var + 1e-6f);
#pragma unroll
    for (int i = 0; i < 2; ++i) {
        int c = t + i * 256;
        float y = (v[i] - mean) * inv * w[c] + bias[c];
        if (outf) outf[(size_t)row * C_ + c] = y;
        if (outb) outb[(size_t)row * C_ + c] = (bf16)y;
    }
}

// ---------------- GEMM: Out = epilogue(A[M,LDA] * W[N,LDW]^T + bias) ----------------
// Block tile 256x64 (8 waves x 32 rows), K chunks of 64, async double-buffered LDS.
template <int LDA, int LDW, int NK, bool GELU>
__global__ __launch_bounds__(256) void gemm_kernel(const bf16* __restrict__ A,
                                                   const bf16* __restrict__ W,
                                                   const float* __restrict__ bias,
                                                   const float* __restrict__ resid,
                                                   float* __restrict__ outf,
                                                   bf16* __restrict__ outb,
                                                   int out_ld) {
    int mb = blockIdx.x * 256, nb = blockIdx.y * 64;
    int tid = threadIdx.x, wave = tid >> 5, lane = tid & 31;
    __shared__ alignas(16) bf16 at[2][256 * 64];
    __shared__ alignas(16) bf16 wt[2][64 * 64];
    unsigned at_lds[2] = {(unsigned)(uintptr_t)&at[0][0], (unsigned)(uintptr_t)&at[1][0]};
    unsigned wt_lds[2] = {(unsigned)(uintptr_t)&wt[0][0], (unsigned)(uintptr_t)&wt[1][0]};
    const bf16* Ab = A + (size_t)mb * LDA;
    const bf16* Wb = W + (size_t)nb * LDW;

    auto issue = [&](int chunk, int buf) {
        int kc = chunk * 64;
#pragma unroll
        for (int j = 0; j < 8; ++j) {       // A tile: 2048 16B vectors / 256 threads
            int i = tid + j * 256;
            int row = i >> 3, c8 = (i & 7) * 8;
            async_b128(at_lds[buf] + (unsigned)(row * 64 + c8) * 2,
                       (unsigned)((row * LDA + kc + c8) * 2), Ab);
        }
#pragma unroll
        for (int j = 0; j < 2; ++j) {       // W tile: 512 16B vectors / 256 threads
            int i = tid + j * 256;
            int row = i >> 3, c8 = (i & 7) * 8;
            async_b128(wt_lds[buf] + (unsigned)(row * 64 + c8) * 2,
                       (unsigned)((row * LDW + kc + c8) * 2), Wb);
        }
    };

    floatx8 acc[8] = {};
    issue(0, 0);
    for (int c = 0; c < NK; ++c) {
        if (c + 1 < NK) {
            issue(c + 1, (c + 1) & 1);
            asm volatile("s_wait_asynccnt 0xA" ::: "memory");  // chunk c landed (in-order)
        } else {
            asm volatile("s_wait_asynccnt 0x0" ::: "memory");
        }
        __syncthreads();
        const bf16* atile = at[c & 1];
        const bf16* wtile = wt[c & 1];
        bf16x16 a0 = load_a_frag(atile + (wave * 32) * 64, 64, 0, lane);
        bf16x16 a1 = load_a_frag(atile + (wave * 32) * 64, 64, 32, lane);
        bf16x16 a2 = load_a_frag(atile + (wave * 32 + 16) * 64, 64, 0, lane);
        bf16x16 a3 = load_a_frag(atile + (wave * 32 + 16) * 64, 64, 32, lane);
#pragma unroll
        for (int t = 0; t < 4; ++t) {
            bf16x16 b0 = load_b_frag(wtile, 64, t * 16, 0, lane);
            bf16x16 b1 = load_b_frag(wtile, 64, t * 16, 32, lane);
            acc[t]     = wmma_bf16(a0, b0, acc[t]);
            acc[t]     = wmma_bf16(a1, b1, acc[t]);
            acc[4 + t] = wmma_bf16(a2, b0, acc[4 + t]);
            acc[4 + t] = wmma_bf16(a3, b1, acc[4 + t]);
        }
        __syncthreads();   // all waves done with buf (c&1) before it is overwritten
    }

    int row0 = mb + wave * 32;
#pragma unroll
    for (int u = 0; u < 2; ++u) {
#pragma unroll
        for (int t = 0; t < 4; ++t) {
            int n = nb + t * 16 + (lane & 15);
            float bj = bias[n];
#pragma unroll
            for (int r = 0; r < 8; ++r) {
                int m = row0 + u * 16 + r + (lane >> 4) * 8;
                float x = acc[u * 4 + t][r] + bj;
                if (GELU) {
                    float g = 0.5f * x * (1.0f + erff(x * 0.70710678118654752f));
                    outb[(size_t)m * out_ld + n] = (bf16)g;
                } else {
                    outf[(size_t)m * out_ld + n] = x + resid[(size_t)m * out_ld + n];
                }
            }
        }
    }
}

extern "C" void kernel_launch(void* const* d_in, const int* in_sizes, int n_in,
                              void* d_out, int out_size, void* d_ws, size_t ws_size,
                              hipStream_t stream) {
    const float* q     = (const float*)d_in[0];
    const float* k     = (const float*)d_in[1];
    const float* v     = (const float*)d_in[2];
    const float* fc1_w = (const float*)d_in[3];
    const float* fc1_b = (const float*)d_in[4];
    const float* fc2_w = (const float*)d_in[5];
    const float* fc2_b = (const float*)d_in[6];
    const float* ln1_w = (const float*)d_in[7];
    const float* ln1_b = (const float*)d_in[8];
    const float* ln2_w = (const float*)d_in[9];
    const float* ln2_b = (const float*)d_in[10];
    float* out = (float*)d_out;

    const size_t QKV = (size_t)B_ * N_ * C_;        // 4,194,304
    const size_t WSZ = (size_t)HIDE_ * C_;          // 1,048,576
    const size_t ROWS = (size_t)B_ * N_;            // 8192

    char* ws = (char*)d_ws;
    size_t off = 0;
    bf16* qb  = (bf16*)(ws + off); off += QKV * 2;
    bf16* kb  = (bf16*)(ws + off); off += QKV * 2;
    bf16* vb  = (bf16*)(ws + off); off += QKV * 2;
    bf16* w1b = (bf16*)(ws + off); off += WSZ * 2;
    bf16* w2b = (bf16*)(ws + off); off += WSZ * 2;
    float* attn   = (float*)(ws + off); off += QKV * 4;
    float* q2f    = (float*)(ws + off); off += QKV * 4;
    bf16*  q2b    = (bf16*)(ws + off);  off += QKV * 2;
    bf16*  hbuf   = (bf16*)(ws + off);  off += ROWS * HIDE_ * 2;
    float* mlpres = (float*)(ws + off); off += QKV * 4;
    (void)ws_size; (void)in_sizes; (void)n_in; (void)out_size;

    {   // fp32 -> bf16 (q pre-scaled by 1/sqrt(hd)=0.125)
        int n4q = (int)(QKV / 4), n4w = (int)(WSZ / 4);
        cvt_kernel<<<(n4q + 255) / 256, 256, 0, stream>>>(q, qb, 0.125f, n4q);
        cvt_kernel<<<(n4q + 255) / 256, 256, 0, stream>>>(k, kb, 1.0f, n4q);
        cvt_kernel<<<(n4q + 255) / 256, 256, 0, stream>>>(v, vb, 1.0f, n4q);
        cvt_kernel<<<(n4w + 255) / 256, 256, 0, stream>>>(fc1_w, w1b, 1.0f, n4w);
        cvt_kernel<<<(n4w + 255) / 256, 256, 0, stream>>>(fc2_w, w2b, 1.0f, n4w);
    }
    attn_kernel<<<dim3(N_ / 64, H_, B_), 128, 0, stream>>>(qb, kb, vb, attn);
    ln_kernel<<<(int)ROWS, 256, 0, stream>>>(q, attn, ln1_w, ln1_b, q2f, q2b);
    gemm_kernel<C_, C_, C_ / 64, true>
        <<<dim3(ROWS / 256, HIDE_ / 64), 256, 0, stream>>>(q2b, w1b, fc1_b, nullptr,
                                                           nullptr, hbuf, HIDE_);
    gemm_kernel<HIDE_, HIDE_, HIDE_ / 64, false>
        <<<dim3(ROWS / 256, C_ / 64), 256, 0, stream>>>(hbuf, w2b, fc2_b, q2f,
                                                        mlpres, nullptr, C_);
    ln_kernel<<<(int)ROWS, 256, 0, stream>>>(mlpres, nullptr, ln2_w, ln2_b, out, nullptr);
}
